// GroupedQueryAttention_43911745634334
// MI455X (gfx1250) — matrix operand order repaired and checked by reading, not compiled, not run on hardware
//
#include <hip/hip_runtime.h>
#include <hip/hip_bf16.h>
#include <math.h>

typedef __attribute__((ext_vector_type(16))) _Float16 v16h;
typedef __attribute__((ext_vector_type(8)))  _Float16 v8h;
typedef __attribute__((ext_vector_type(4)))  _Float16 v4h;
typedef __attribute__((ext_vector_type(8)))  float    v8f;

#define B_      2
#define S_      2048
#define HID_    2048
#define NH_     32
#define NKV_    8
#define HD_     64
#define EPS_    1e-6f
#define SCALE_  0.125f   // 64^-0.5, folded into Q

static __device__ inline v16h make16(v8h lo, v8h hi) {
  v16h r;
#pragma unroll
  for (int i = 0; i < 8; ++i) { r[i] = lo[i]; r[i + 8] = hi[i]; }
  return r;
}

static __device__ inline v8f wmma_f16(v16h a, v16h b, v8f c) {
  return __builtin_amdgcn_wmma_f32_16x16x32_f16(false, a, false, b, (short)0, c,
                                                false, false);
}

// ---------------------------------------------------------------------------
// C[M,N] = A[M,K] * W[N,K]^T   (A, W, C all f32 row-major; f16 WMMA inside)
// 64x64 tile per 128-thread block; K stepped by 32.
// Each thread stages 16 consecutive f32 per operand: 4x global_load_b128,
// packed cvt, 2x ds_store_b128.
// ---------------------------------------------------------------------------
__global__ __launch_bounds__(128) void gemm_xwT(const float* __restrict__ A,
                                                const float* __restrict__ W,
                                                float* __restrict__ C,
                                                int M, int N, int K) {
  __shared__ _Float16 As[64 * 32];
  __shared__ _Float16 Bs[64 * 32];
  const int tid = threadIdx.x;
  const int lane = tid & 31, wave = tid >> 5;
  const int laneLo = lane & 15, laneHi = lane >> 4;
  const int m0 = blockIdx.x * 64, n0 = blockIdx.y * 64;
  v8f acc[4] = {};

  const int lr = tid >> 1;             // 0..63: tile row this thread stages
  const int lc = (tid & 1) * 16;       // 0 or 16: k-offset within tile
  const float* pA = A + (size_t)(m0 + lr) * K + lc;
  const float* pW = W + (size_t)(n0 + lr) * K + lc;

  for (int k0 = 0; k0 < K; k0 += 32) {
    float4 av[4], bv[4];
#pragma unroll
    for (int i = 0; i < 4; ++i) av[i] = *(const float4*)(pA + k0 + i * 4);
#pragma unroll
    for (int i = 0; i < 4; ++i) bv[i] = *(const float4*)(pW + k0 + i * 4);
    if (k0 + 32 < K) {
      __builtin_prefetch(pA + k0 + 32, 0, 1);
      __builtin_prefetch(pW + k0 + 32, 0, 1);
    }
    v8h ah0, ah1, bh0, bh1;
#pragma unroll
    for (int i = 0; i < 8; ++i) {
      ah0[i] = (_Float16)(((const float*)av)[i]);
      ah1[i] = (_Float16)(((const float*)av)[i + 8]);
      bh0[i] = (_Float16)(((const float*)bv)[i]);
      bh1[i] = (_Float16)(((const float*)bv)[i + 8]);
    }
    __syncthreads();  // previous iteration's fragment reads are done
    *(v8h*)&As[tid * 16]     = ah0;
    *(v8h*)&As[tid * 16 + 8] = ah1;
    *(v8h*)&Bs[tid * 16]     = bh0;
    *(v8h*)&Bs[tid * 16 + 8] = bh1;
    __syncthreads();  // tiles visible to all waves

    // A fragment (16x32): rows wave*16+laneLo, k = 16*laneHi + i (16 contiguous halves)
    const _Float16* ap = &As[(wave * 16 + laneLo) * 32 + laneHi * 16];
    v16h afrag = make16(*(const v8h*)ap, *(const v8h*)(ap + 8));
#pragma unroll
    for (int f = 0; f < 4; ++f) {
      // B fragment (32x16): n = laneLo, k = 16*laneHi + i (16 contiguous halves)
      const _Float16* bp = &Bs[(f * 16 + laneLo) * 32 + laneHi * 16];
      acc[f] = wmma_f16(afrag, *(const v16h*)bp, acc[f]);
    }
  }
#pragma unroll
  for (int f = 0; f < 4; ++f)
#pragma unroll
    for (int r = 0; r < 8; ++r) {
      int m = m0 + wave * 16 + r + laneHi * 8;
      int n = n0 + f * 16 + laneLo;
      C[(size_t)m * N + n] = acc[f][r];
    }
}

// ---------------------------------------------------------------------------
// Per-head RMSNorm + RoPE + f16 repack. One wave per (b,s,head-slot).
// Slots: [0,32) Q heads, [32,40) K heads, [40,48) V heads.
// ---------------------------------------------------------------------------
__global__ __launch_bounds__(128) void norm_rope(
    const float* __restrict__ Q, const float* __restrict__ Kk,
    const float* __restrict__ V, const float* __restrict__ cosT,
    const float* __restrict__ sinT, const float* __restrict__ qw,
    const float* __restrict__ kw, _Float16* __restrict__ Qh,
    _Float16* __restrict__ Kh, _Float16* __restrict__ Vh) {
  const int lane = threadIdx.x & 31;
  const int wgid = blockIdx.x * 4 + (threadIdx.x >> 5);
  const int total = B_ * S_ * (NH_ + 2 * NKV_);
  if (wgid >= total) return;
  const int slot = wgid % 48;
  const int bs = wgid / 48;
  const int s = bs % S_, b = bs / S_;

  if (slot < NH_ + NKV_) {
    const bool isQ = slot < NH_;
    const int h = isQ ? slot : slot - NH_;
    const float* src = isQ ? Q + (size_t)(b * S_ + s) * 2048 + h * 64
                           : Kk + (size_t)(b * S_ + s) * 512 + h * 64;
    const float* w = isQ ? qw : kw;
    float x0 = src[lane], x1 = src[lane + 32];
    float ss = x0 * x0 + x1 * x1;
#pragma unroll
    for (int m = 1; m < 32; m <<= 1) ss += __shfl_xor(ss, m, 32);
    float rinv = rsqrtf(ss * (1.0f / 64.0f) + EPS_);
    float n0 = x0 * rinv * w[lane];          // dims 0..31 (RoPE'd)
    float n1 = x1 * rinv * w[lane + 32];     // dims 32..63 (pass-through)
    float part = __shfl_xor(n0, 16, 32);
    float rot = (lane < 16) ? -part : part;  // rotate_half on 32-dim block
    float c = cosT[(size_t)(b * S_ + s) * 32 + lane];
    float sn = sinT[(size_t)(b * S_ + s) * 32 + lane];
    float r0 = n0 * c + rot * sn;
    float scl = isQ ? SCALE_ : 1.0f;
    _Float16* dst = isQ ? Qh + (((size_t)b * NH_ + h) * S_ + s) * 64
                        : Kh + (((size_t)b * NKV_ + h) * S_ + s) * 64;
    dst[lane] = (_Float16)(r0 * scl);
    dst[lane + 32] = (_Float16)(n1 * scl);
  } else {
    const int kv = slot - (NH_ + NKV_);
    const float* src = V + (size_t)(b * S_ + s) * 512 + kv * 64;
    _Float16* dst = Vh + (((size_t)b * NKV_ + kv) * S_ + s) * 64;
    dst[lane] = (_Float16)src[lane];
    dst[lane + 32] = (_Float16)src[lane + 32];
  }
}

// ---------------------------------------------------------------------------
// Flash-style causal GQA attention. Block = 128 threads = 4 waves handles one
// 64-query tile of one head. Output written as (B, S, NH*HD) f32.
// ---------------------------------------------------------------------------
__global__ __launch_bounds__(128) void attn(const _Float16* __restrict__ Qh,
                                            const _Float16* __restrict__ Kh,
                                            const _Float16* __restrict__ Vh,
                                            float* __restrict__ O) {
  __shared__ _Float16 Ks[64 * 64];          // [key][d]
  __shared__ _Float16 VsT[64 * 64];         // [d][key] (transposed)
  __shared__ _Float16 Ps[4][16 * 64];       // per-wave P tile
  const int tid = threadIdx.x;
  const int lane = tid & 31, wave = tid >> 5;
  const int laneLo = lane & 15, laneHi = lane >> 4;
  const int qt = blockIdx.x, h = blockIdx.y, b = blockIdx.z;
  const int kv = h >> 2;                    // GROUPS = 4
  const int qBase = qt * 64;
  const _Float16* Qp = Qh + (((size_t)b * NH_ + h) * S_ + qBase) * 64;
  const _Float16* Kp = Kh + (((size_t)b * NKV_ + kv) * S_) * 64;
  const _Float16* Vp = Vh + (((size_t)b * NKV_ + kv) * S_) * 64;

  // Q fragments (constant across key tiles): rows wave*16+laneLo, two K-chunks
  v16h qf[2];
  {
    const _Float16* qrow = Qp + (wave * 16 + laneLo) * 64 + laneHi * 16;
#pragma unroll
    for (int c = 0; c < 2; ++c)
      qf[c] = make16(*(const v8h*)(qrow + c * 32),
                     *(const v8h*)(qrow + c * 32 + 8));
  }
  v8f oacc[4] = {};
  float mrow[8], lrow[8];
#pragma unroll
  for (int r = 0; r < 8; ++r) { mrow[r] = -1e30f; lrow[r] = 0.0f; }

  // staging coordinates
  const int lr = tid >> 1, lc = (tid & 1) * 32;     // K tile: 32 contiguous halves
  const int dblk = (tid & 15) * 4, tblk = (tid >> 4) * 8;  // V: 8(t)x4(d) block

  for (int kt = 0; kt <= qt; ++kt) {
    const _Float16* kbase = Kp + (size_t)kt * 64 * 64;
    const _Float16* vbase = Vp + (size_t)kt * 64 * 64;

    // ---- stage K tile (row-major) ----
    v8h kt0 = *(const v8h*)(kbase + lr * 64 + lc);
    v8h kt1 = *(const v8h*)(kbase + lr * 64 + lc + 8);
    v8h kt2 = *(const v8h*)(kbase + lr * 64 + lc + 16);
    v8h kt3 = *(const v8h*)(kbase + lr * 64 + lc + 24);
    // ---- stage V tile (transposed, 8x4 sub-blocks) ----
    v4h rowv[8];
#pragma unroll
    for (int i = 0; i < 8; ++i)
      rowv[i] = *(const v4h*)(vbase + (tblk + i) * 64 + dblk);
    if (kt < qt) {
      __builtin_prefetch(kbase + 64 * 64 + lr * 64 + lc, 0, 1);
      __builtin_prefetch(vbase + 64 * 64 + tblk * 64 + dblk, 0, 1);
    }
    __syncthreads();  // previous iteration's LDS reads complete
    {
      _Float16* dstp = &Ks[lr * 64 + lc];
      *(v8h*)(dstp) = kt0;
      *(v8h*)(dstp + 8) = kt1;
      *(v8h*)(dstp + 16) = kt2;
      *(v8h*)(dstp + 24) = kt3;
    }
#pragma unroll
    for (int j = 0; j < 4; ++j) {
      v8h col;
#pragma unroll
      for (int i = 0; i < 8; ++i) col[i] = rowv[i][j];
      *(v8h*)&VsT[(dblk + j) * 64 + tblk] = col;
    }
    __syncthreads();

    // S = Q * K^T  (scale pre-folded into Q)
    v8f sf[4] = {};
#pragma unroll
    for (int f = 0; f < 4; ++f)
#pragma unroll
      for (int c = 0; c < 2; ++c) {
        const _Float16* bp = &Ks[(f * 16 + laneLo) * 64 + c * 32 + laneHi * 16];
        sf[f] = wmma_f16(qf[c], *(const v16h*)bp, sf[f]);
      }

    if (kt == qt) {  // diagonal tile: causal mask (tiles are 64-aligned)
#pragma unroll
      for (int f = 0; f < 4; ++f)
#pragma unroll
        for (int r = 0; r < 8; ++r) {
          int m = wave * 16 + r + laneHi * 8;
          int n = f * 16 + laneLo;
          sf[f][r] = (n > m) ? -1e30f : sf[f][r];
        }
    }

    // Online softmax. Row r lives in lanes of this half (xor masks < 16).
    _Float16* pw = &Ps[wave][0];
#pragma unroll
    for (int r = 0; r < 8; ++r) {
      float mx = fmaxf(fmaxf(sf[0][r], sf[1][r]), fmaxf(sf[2][r], sf[3][r]));
#pragma unroll
      for (int msk = 1; msk < 16; msk <<= 1)
        mx = fmaxf(mx, __shfl_xor(mx, msk, 32));
      float nm = fmaxf(mrow[r], mx);
      float alpha = __expf(mrow[r] - nm);
      mrow[r] = nm;
      float rs = 0.0f;
#pragma unroll
      for (int f = 0; f < 4; ++f) {
        float p = __expf(sf[f][r] - nm);
        rs += p;
        pw[(r + laneHi * 8) * 64 + f * 16 + laneLo] = (_Float16)p;
      }
#pragma unroll
      for (int msk = 1; msk < 16; msk <<= 1) rs += __shfl_xor(rs, msk, 32);
      lrow[r] = lrow[r] * alpha + rs;
#pragma unroll
      for (int f = 0; f < 4; ++f) oacc[f][r] = oacc[f][r] * alpha;
    }

    // O += P * V  (P via per-wave LDS round-trip; V read transposed)
#pragma unroll
    for (int c = 0; c < 2; ++c) {
      const _Float16* ap = &pw[laneLo * 64 + c * 32 + laneHi * 16];
      v16h af = make16(*(const v8h*)ap, *(const v8h*)(ap + 8));
#pragma unroll
      for (int f = 0; f < 4; ++f) {
        const _Float16* bp = &VsT[(f * 16 + laneLo) * 64 + c * 32 + laneHi * 16];
        oacc[f] = wmma_f16(af, *(const v16h*)bp, oacc[f]);
      }
    }
  }

#pragma unroll
  for (int f = 0; f < 4; ++f)
#pragma unroll
    for (int r = 0; r < 8; ++r) {
      int s = qBase + wave * 16 + r + laneHi * 8;
      int col = h * 64 + f * 16 + laneLo;
      O[((size_t)(b * S_ + s)) * 2048 + col] = oacc[f][r] / lrow[r];
    }
}

// ---------------------------------------------------------------------------
// out *= sigmoid(gate), in place.
// ---------------------------------------------------------------------------
__global__ __launch_bounds__(256) void gate_mul(float* __restrict__ O,
                                                const float* __restrict__ G,
                                                int n) {
  int i = blockIdx.x * 256 + threadIdx.x;
  if (i < n) {
    float g = G[i];
    O[i] = O[i] * (1.0f / (1.0f + __expf(-g)));
  }
}

extern "C" void kernel_launch(void* const* d_in, const int* in_sizes, int n_in,
                              void* d_out, int out_size, void* d_ws,
                              size_t ws_size, hipStream_t stream) {
  (void)in_sizes; (void)n_in; (void)out_size; (void)ws_size;
  const float* hidden = (const float*)d_in[0];
  const float* cosT   = (const float*)d_in[1];
  const float* sinT   = (const float*)d_in[2];
  const float* Wq     = (const float*)d_in[3];
  const float* Wk     = (const float*)d_in[4];
  const float* Wv     = (const float*)d_in[5];
  const float* Wg     = (const float*)d_in[6];
  const float* Wo     = (const float*)d_in[7];
  const float* qw     = (const float*)d_in[8];
  const float* kw     = (const float*)d_in[9];
  float* out = (float*)d_out;

  char* ws = (char*)d_ws;
  float*    Qf = (float*)(ws);                              // 32 MiB
  float*    Kf = (float*)(ws + ((size_t)32 << 20));         //  8 MiB
  float*    Vf = (float*)(ws + ((size_t)40 << 20));         //  8 MiB
  float*    Gf = (float*)(ws + ((size_t)48 << 20));         // 32 MiB
  _Float16* Qh = (_Float16*)(ws + ((size_t)80 << 20));      // 16 MiB
  _Float16* Kh = (_Float16*)(ws + ((size_t)96 << 20));      //  4 MiB
  _Float16* Vh = (_Float16*)(ws + ((size_t)100 << 20));     //  4 MiB
  float*    Of = Qf;  // attention output reuses dead Q-f32 region

  const int M = B_ * S_;
  dim3 blk(128);
  gemm_xwT<<<dim3(M / 64, HID_ / 64), blk, 0, stream>>>(hidden, Wq, Qf, M, HID_, HID_);
  gemm_xwT<<<dim3(M / 64, 512 / 64), blk, 0, stream>>>(hidden, Wk, Kf, M, 512, HID_);
  gemm_xwT<<<dim3(M / 64, 512 / 64), blk, 0, stream>>>(hidden, Wv, Vf, M, 512, HID_);
  gemm_xwT<<<dim3(M / 64, HID_ / 64), blk, 0, stream>>>(hidden, Wg, Gf, M, HID_, HID_);

  const int waves = B_ * S_ * (NH_ + 2 * NKV_);
  norm_rope<<<dim3((waves + 3) / 4), blk, 0, stream>>>(Qf, Kf, Vf, cosT, sinT,
                                                       qw, kw, Qh, Kh, Vh);

  attn<<<dim3(S_ / 64, NH_, B_), blk, 0, stream>>>(Qh, Kh, Vh, Of);

  const int n = B_ * S_ * HID_;
  gate_mul<<<dim3(n / 256), dim3(256), 0, stream>>>(Of, Gf, n);

  gemm_xwT<<<dim3(M / 64, HID_ / 64), blk, 0, stream>>>(Of, Wo, out, M, HID_, HID_);
}
